// SeqUnit_8735963480663
// MI455X (gfx1250) — matrix operand order repaired and checked by reading, not compiled, not run on hardware
//
#include <hip/hip_runtime.h>
#include <cstdint>
#include <cstddef>

// ---------------- problem constants ----------------
#define BB 32
#define SS 100
#define TT 50
#define HH 500
#define EE 400
#define FSS 50
#define PP 5
#define VV 20003
#define UNI 460
#define FENC 60
// padded K dims (multiples of 32)
#define UNIP 480
#define FENCP 64
#define HP 512
#define EP 416
#define H2P 1024
#define G4 2000
#define G2 1000

typedef __attribute__((ext_vector_type(16))) __bf16 v16bf;
typedef __attribute__((ext_vector_type(8)))  __bf16 v8bf;
typedef __attribute__((ext_vector_type(8)))  float  v8f;

// bf16 stored as unsigned short in buffers (host-portable), reinterpreted on device.
__device__ __forceinline__ unsigned short f2bf(float f) {
    union { float f; unsigned u; } x; x.f = f;
    unsigned r = x.u + 0x7FFFu + ((x.u >> 16) & 1u);
    return (unsigned short)(r >> 16);
}
__device__ __forceinline__ float sigf(float x) { return 1.0f / (1.0f + __expf(-x)); }

#define WMMA_BF16(A_, B_, C_) \
    __builtin_amdgcn_wmma_f32_16x16x32_bf16(false, (A_), false, (B_), (short)0, (C_), false, false)

// ---------------- generic WMMA bf16 GEMM ----------------
// C[M,N](f32) = A[M,K](bf16 row-major, lda mult of 8) * Bpk (fragment-packed bf16)
//               [+bias[N]] [+add[M,N]] [tanh] ; optional bf16 mirror Cb.
// Bpk layout: element ((nt*KT + kt)*32 + lane)*16 + i  where
//   kk = kt*32 + (i&8)*2 + (lane>>4)*8 + (i&7) , nn = nt*16 + (lane&15)
// One wave: one 16-row M tile x two 16-col N tiles; 2-stage software pipeline.
// grid = (ceil(NT/2), ceil(M/16)), block = 32.
__global__ __launch_bounds__(32) void gemm_bf16(
    const unsigned short* __restrict__ Au, int lda,
    const unsigned short* __restrict__ Bu,
    float* __restrict__ C, int ldc,
    const float* __restrict__ bias,
    const float* __restrict__ add, int ldadd,
    unsigned short* __restrict__ Cbu, int ldcb,
    int M, int N, int KT, int NT, int act)
{
    const int lane = threadIdx.x & 31;
    const int half = lane >> 4;       // 0 or 1
    const int l15  = lane & 15;
    const int tm  = blockIdx.y * 16;
    const int nt0 = blockIdx.x * 2;
    const int nt1 = nt0 + 1;
    const bool has1 = (nt1 < NT);

    int am = tm + l15; if (am >= M) am = M - 1;   // clamp; masked rows not stored
    const v8bf* __restrict__ arow =
        reinterpret_cast<const v8bf*>(Au + (size_t)am * lda);
    const v8bf* __restrict__ bbase = reinterpret_cast<const v8bf*>(Bu);
    const v8bf* __restrict__ b0p = bbase + ((size_t)nt0 * KT * 32 + lane) * 2;
    const v8bf* __restrict__ b1p = has1 ? bbase + ((size_t)nt1 * KT * 32 + lane) * 2
                                        : b0p;

    union V16 { v16bf v; v8bf h[2]; };
    v8f acc0 = {};
    v8f acc1 = {};

    // preload k-tile 0
    V16 a, b0, b1;
    a.h[0] = arow[half];
    a.h[1] = arow[half + 2];
    b0.h[0] = b0p[0];
    b0.h[1] = b0p[1];
    b1 = b0;

    if (has1) {
        b1.h[0] = b1p[0];
        b1.h[1] = b1p[1];
        for (int kt = 0; kt < KT - 1; ++kt) {
            V16 an, b0n, b1n;                       // fetch k-tile kt+1
            an.h[0]  = arow[(kt + 1) * 4 + half];
            an.h[1]  = arow[(kt + 1) * 4 + half + 2];
            b0n.h[0] = b0p[64];
            b0n.h[1] = b0p[65];
            b1n.h[0] = b1p[64];
            b1n.h[1] = b1p[65];
            __builtin_prefetch((const char*)b0p + 4096, 0, 1);   // global_prefetch_b8
            __builtin_prefetch((const char*)b1p + 4096, 0, 1);
            acc0 = WMMA_BF16(a.v, b0.v, acc0);      // overlaps the 6 in-flight loads
            acc1 = WMMA_BF16(a.v, b1.v, acc1);
            a = an; b0 = b0n; b1 = b1n;
            b0p += 64; b1p += 64;
        }
        acc0 = WMMA_BF16(a.v, b0.v, acc0);
        acc1 = WMMA_BF16(a.v, b1.v, acc1);
    } else {
        for (int kt = 0; kt < KT - 1; ++kt) {
            V16 an, b0n;
            an.h[0]  = arow[(kt + 1) * 4 + half];
            an.h[1]  = arow[(kt + 1) * 4 + half + 2];
            b0n.h[0] = b0p[64];
            b0n.h[1] = b0p[65];
            __builtin_prefetch((const char*)b0p + 4096, 0, 1);
            acc0 = WMMA_BF16(a.v, b0.v, acc0);
            a = an; b0 = b0n;
            b0p += 64;
        }
        acc0 = WMMA_BF16(a.v, b0.v, acc0);
    }

    // C/D layout: VGPR r -> row = tm + half*8 + r, col = ntile*16 + (lane&15)
    const int rowbase = tm + half * 8;
    for (int tsel = 0; tsel < 2; ++tsel) {
        if (tsel == 1 && !has1) break;
        const v8f& acc = tsel ? acc1 : acc0;
        const int col = (nt0 + tsel) * 16 + l15;
        if (col < N) {
            const float bz = bias ? bias[col] : 0.0f;
#pragma unroll
            for (int r = 0; r < 8; ++r) {
                const int row = rowbase + r;
                if (row < M) {
                    float v = acc[r] + bz;
                    if (add) v += add[(size_t)row * ldadd + col];
                    if (act) v = tanhf(v);
                    if (C)   C[(size_t)row * ldc + col] = v;
                    if (Cbu) Cbu[(size_t)row * ldcb + col] = f2bf(v);
                }
            }
        }
    }
}

// ---------------- weight pack (f32 -> bf16 fragment order, zero-padded) ----
// transposed=0: src is [K,N] row-major ; transposed=1: src is [N,K] (torch [out,in])
__global__ void pack_b(const float* __restrict__ src, unsigned short* __restrict__ dst,
                       int K, int N, int KT, int NT, int transposed)
{
    int idx = blockIdx.x * blockDim.x + threadIdx.x;
    if (idx >= NT * KT * 512) return;
    const int i    = idx & 15;
    const int lane = (idx >> 4) & 31;
    const int kt   = (idx >> 9) % KT;
    const int nt   = (idx >> 9) / KT;
    const int kk = kt * 32 + (i < 8 ? (lane >> 4) * 8 + i : 16 + (lane >> 4) * 8 + (i - 8));
    const int nn = nt * 16 + (lane & 15);
    float v = 0.0f;
    if (kk < K && nn < N)
        v = transposed ? src[(size_t)nn * K + kk] : src[(size_t)kk * N + nn];
    dst[idx] = f2bf(v);
}

// ---------------- embedding / concat ----------------
// word row r = s*B + b : [enc_emb | field | pos | rpos | 0pad] -> bf16 [3200,UNIP]
// fin  row r           : [field | pos | rpos | 0pad]           -> bf16 [3200,FENCP]
__global__ void embed_concat(const int* __restrict__ text, const int* __restrict__ field,
                             const int* __restrict__ pos, const int* __restrict__ rpos,
                             const float* __restrict__ ee, const float* __restrict__ fe,
                             const float* __restrict__ pe, const float* __restrict__ re,
                             unsigned short* __restrict__ wordB, unsigned short* __restrict__ finB)
{
    const int r = blockIdx.x;                 // 0..3199
    const int s = r / BB, b = r % BB;
    const int ti = text[b * SS + s];
    const int fi = field[b * SS + s];
    const int pi = pos[b * SS + s];
    const int ri = rpos[b * SS + s];
    for (int j = threadIdx.x; j < UNIP; j += blockDim.x) {
        float v = 0.0f;
        if (j < EE)                  v = ee[(size_t)ti * EE + j];
        else if (j < EE + FSS)       v = fe[(size_t)fi * FSS + (j - EE)];
        else if (j < EE + FSS + PP)  v = pe[(size_t)pi * PP + (j - EE - FSS)];
        else if (j < UNI)            v = re[(size_t)ri * PP + (j - EE - FSS - PP)];
        wordB[(size_t)r * UNIP + j] = f2bf(v);
        const int fj = j - EE;
        if (fj >= 0 && fj < FENCP)
            finB[(size_t)r * FENCP + fj] = f2bf(j < UNI ? v : 0.0f);
    }
}

// decoder teacher-forced inputs: row r = t*B + b -> bf16 [1600,EP]
__global__ void dec_embed(const int* __restrict__ target, const float* __restrict__ ee,
                          unsigned short* __restrict__ decXB) {
    const int r = blockIdx.x;                 // 0..1599
    const int t = r / BB, b = r % BB;
    const int ti = target[b * TT + t];
    for (int j = threadIdx.x; j < EP; j += blockDim.x)
        decXB[(size_t)r * EP + j] = f2bf(j < EE ? ee[(size_t)ti * EE + j] : 0.0f);
}

// xwxp[t*B+b] = enc_emb[target[b,t]] . W_xp   (one wave per row)
__global__ void xwxp_kernel(const int* __restrict__ target, const float* __restrict__ ee,
                            const float* __restrict__ Wxp, float* __restrict__ xwxp) {
    const int wid = (blockIdx.x * blockDim.x + threadIdx.x) >> 5;
    const int lane = threadIdx.x & 31;
    if (wid >= TT * BB) return;
    const int t = wid / BB, b = wid % BB;
    const int ti = target[b * TT + t];
    float acc = 0.0f;
    for (int j = lane; j < EE; j += 32) acc += ee[(size_t)ti * EE + j] * Wxp[j];
    for (int o = 16; o; o >>= 1) acc += __shfl_down(acc, o);
    if (!lane) xwxp[wid] = acc;
}

// ---------------- LSTM gate kernels ----------------
// encoder: g[32,2000] already = x@W_f + h@U_f + bias_f ; zz row = (s*B+b) of ZW (z@Wz+biasz)
__global__ void enc_gate(const float* __restrict__ g, const float* __restrict__ zz,
                         float* __restrict__ h, float* __restrict__ c,
                         unsigned short* __restrict__ hB,
                         float* __restrict__ en_out, unsigned short* __restrict__ enB, int s)
{
    int idx = blockIdx.x * blockDim.x + threadIdx.x;
    if (idx >= BB * HH) return;
    const int b = idx / HH, j = idx - b * HH;
    const float* gr = g + (size_t)b * G4;
    const float* zr = zz + (size_t)(s * BB + b) * G2;
    const float i  = sigf(gr[j]);
    const float f  = sigf(gr[HH + j] + 1.0f);
    const float gg = tanhf(gr[2 * HH + j]);
    const float o  = sigf(gr[3 * HH + j]);
    const float l  = sigf(zr[j]);
    const float zt = tanhf(zr[HH + j]);
    const float cv = f * c[idx] + i * gg + l * zt;
    c[idx] = cv;
    const float hv = o * tanhf(cv);
    h[idx] = hv;
    hB[(size_t)b * HP + j] = f2bf(hv);
    const size_t r = (size_t)(s * BB + b);
    en_out[r * HH + j] = hv;
    enB[r * HP + j] = f2bf(hv);
}

// decoder: plain LSTM (no field gate)
__global__ void dec_gate(const float* __restrict__ g, float* __restrict__ h,
                         float* __restrict__ c, unsigned short* __restrict__ hB)
{
    int idx = blockIdx.x * blockDim.x + threadIdx.x;
    if (idx >= BB * HH) return;
    const int b = idx / HH, j = idx - b * HH;
    const float* gr = g + (size_t)b * G4;
    const float i  = sigf(gr[j]);
    const float f  = sigf(gr[HH + j] + 1.0f);
    const float gg = tanhf(gr[2 * HH + j]);
    const float o  = sigf(gr[3 * HH + j]);
    const float cv = f * c[idx] + i * gg;
    c[idx] = cv;
    const float hv = o * tanhf(cv);
    h[idx] = hv;
    hB[(size_t)b * HP + j] = f2bf(hv);
}

// ---------------- attention (fused scores + softmax-product) ----------------
// one block per b, 128 threads; thread tid handles s=tid (<100):
//   ga = phi[s*B+b].dq[b], gb = psi[s*B+b].dy[b]
//   w  = softmax(ga)*softmax(gb) / (1e-6 + sum)
__global__ void attn_kernel(const float* __restrict__ phi, const float* __restrict__ psi,
                            const float* __restrict__ dq, const float* __restrict__ dy,
                            float* __restrict__ w)
{
    const int b = blockIdx.x, tid = threadIdx.x;   // 128 threads
    __shared__ float red[128];
    __shared__ float mA, sA, mB, sB, sW;
    float va = -1e30f, vb = -1e30f;
    if (tid < SS) {
        const float* ph = phi + ((size_t)tid * BB + b) * HH;
        const float* ps = psi + ((size_t)tid * BB + b) * HH;
        const float* q = dq + (size_t)b * HH;
        const float* y = dy + (size_t)b * HH;
        float sa = 0.0f, sb = 0.0f;
        for (int j = 0; j < HH; ++j) { sa += ph[j] * q[j]; sb += ps[j] * y[j]; }
        va = sa; vb = sb;
    }
    red[tid] = va; __syncthreads();
    for (int o = 64; o; o >>= 1) { if (tid < o) red[tid] = fmaxf(red[tid], red[tid + o]); __syncthreads(); }
    if (!tid) mA = red[0]; __syncthreads();
    const float ea = (tid < SS) ? __expf(va - mA) : 0.0f;
    red[tid] = ea; __syncthreads();
    for (int o = 64; o; o >>= 1) { if (tid < o) red[tid] += red[tid + o]; __syncthreads(); }
    if (!tid) sA = red[0]; __syncthreads();
    red[tid] = vb; __syncthreads();
    for (int o = 64; o; o >>= 1) { if (tid < o) red[tid] = fmaxf(red[tid], red[tid + o]); __syncthreads(); }
    if (!tid) mB = red[0]; __syncthreads();
    const float eb = (tid < SS) ? __expf(vb - mB) : 0.0f;
    red[tid] = eb; __syncthreads();
    for (int o = 64; o; o >>= 1) { if (tid < o) red[tid] += red[tid + o]; __syncthreads(); }
    if (!tid) sB = red[0]; __syncthreads();
    const float wv = (tid < SS) ? (ea / sA) * (eb / sB) : 0.0f;
    red[tid] = wv; __syncthreads();
    for (int o = 64; o; o >>= 1) { if (tid < o) red[tid] += red[tid + o]; __syncthreads(); }
    if (!tid) sW = red[0]; __syncthreads();
    if (tid < SS) w[b * SS + tid] = wv / (1e-6f + sW);
}

// ctx[b,j] = sum_s w[b,s]*en_out[s*B+b, j]; also builds bf16 concat [ctx|h|0] rows
__global__ void ctx_kernel(const float* __restrict__ w, const float* __restrict__ en,
                           const float* __restrict__ h, float* __restrict__ ctx,
                           unsigned short* __restrict__ cat)
{
    int idx = blockIdx.x * blockDim.x + threadIdx.x;
    if (idx >= BB * HH) return;
    const int b = idx / HH, j = idx - b * HH;
    float acc = 0.0f;
    for (int s = 0; s < SS; ++s)
        acc += w[b * SS + s] * en[((size_t)s * BB + b) * HH + j];
    ctx[idx] = acc;
    cat[(size_t)b * H2P + j] = f2bf(acc);
    cat[(size_t)b * H2P + HH + j] = f2bf(h[idx]);
}

// p_gen[b] = sigmoid(ctx.Wc + h.Wd + xwxp + b_p)  (one wave per b)
__global__ void pgen_kernel(const float* __restrict__ ctx, const float* __restrict__ h,
                            const float* __restrict__ Wc, const float* __restrict__ Wd,
                            const float* __restrict__ xwxp, const float* __restrict__ bp,
                            float* __restrict__ pg, int t)
{
    const int wid = (blockIdx.x * blockDim.x + threadIdx.x) >> 5;
    const int lane = threadIdx.x & 31;
    if (wid >= BB) return;
    float acc = 0.0f;
    for (int j = lane; j < HH; j += 32)
        acc += ctx[(size_t)wid * HH + j] * Wc[j] + h[(size_t)wid * HH + j] * Wd[j];
    for (int o = 16; o; o >>= 1) acc += __shfl_down(acc, o);
    if (!lane) pg[wid] = sigf(acc + xwxp[t * BB + wid] + bp[0]);
}

// dist = p_gen * softmax(logits)  -> out[b,t,:]  (one block per b)
__global__ void dist_kernel(const float* __restrict__ logits, const float* __restrict__ pg,
                            float* __restrict__ out, int t)
{
    const int b = blockIdx.x, tid = threadIdx.x;   // 256 threads
    __shared__ float red[256];
    __shared__ float bm, bs;
    const float* l = logits + (size_t)b * VV;
    float m = -1e30f;
    for (int v = tid; v < VV; v += 256) m = fmaxf(m, l[v]);
    red[tid] = m; __syncthreads();
    for (int o = 128; o; o >>= 1) { if (tid < o) red[tid] = fmaxf(red[tid], red[tid + o]); __syncthreads(); }
    if (!tid) bm = red[0]; __syncthreads();
    float s = 0.0f;
    for (int v = tid; v < VV; v += 256) s += __expf(l[v] - bm);
    red[tid] = s; __syncthreads();
    for (int o = 128; o; o >>= 1) { if (tid < o) red[tid] += red[tid + o]; __syncthreads(); }
    if (!tid) bs = red[0]; __syncthreads();
    const float scale = pg[b] / bs;
    float* orow = out + ((size_t)b * TT + t) * VV;
    for (int v = tid; v < VV; v += 256) orow[v] = __expf(l[v] - bm) * scale;
}

// out[b,t,text[b,s]] += (1-p_gen)*w[b,s]  (duplicate tokens -> atomic)
__global__ void scatter_kernel(const float* __restrict__ w, const float* __restrict__ pg,
                               const int* __restrict__ text, float* __restrict__ out, int t)
{
    int idx = blockIdx.x * blockDim.x + threadIdx.x;
    if (idx >= BB * SS) return;
    const int b = idx / SS, s = idx - b * SS;
    atomicAdd(&out[((size_t)b * TT + t) * VV + text[b * SS + s]],
              (1.0f - pg[b]) * w[idx]);
}

// ---------------- host driver ----------------
extern "C" void kernel_launch(void* const* d_in, const int* in_sizes, int n_in,
                              void* d_out, int out_size, void* d_ws, size_t ws_size,
                              hipStream_t stream)
{
    (void)in_sizes; (void)n_in; (void)out_size;
    const int*   text     = (const int*)d_in[0];
    const int*   field    = (const int*)d_in[1];
    const int*   pos      = (const int*)d_in[2];
    const int*   rpos     = (const int*)d_in[3];
    const int*   target   = (const int*)d_in[4];
    const float* enc_emb  = (const float*)d_in[5];
    const float* field_emb= (const float*)d_in[6];
    const float* pos_emb  = (const float*)d_in[7];
    const float* rpos_emb = (const float*)d_in[8];
    const float* W_f      = (const float*)d_in[9];
    const float* Wz       = (const float*)d_in[10];
    const float* U_f      = (const float*)d_in[11];
    const float* bias_f   = (const float*)d_in[12];
    const float* biasz    = (const float*)d_in[13];
    const float* dec_W    = (const float*)d_in[14];
    const float* dec_U    = (const float*)d_in[15];
    const float* dec_bias = (const float*)d_in[16];
    const float* Wp_w     = (const float*)d_in[17];
    const float* Wp_b     = (const float*)d_in[18];
    const float* Wq_w     = (const float*)d_in[19];
    const float* Wq_b     = (const float*)d_in[20];
    const float* Wt_w     = (const float*)d_in[21];
    const float* Wt_b     = (const float*)d_in[22];
    const float* Wx_w     = (const float*)d_in[23];
    const float* Wx_b     = (const float*)d_in[24];
    const float* Wy_w     = (const float*)d_in[25];
    const float* Wy_b     = (const float*)d_in[26];
    const float* out1_w   = (const float*)d_in[27];
    const float* out1_b   = (const float*)d_in[28];
    const float* out2_w   = (const float*)d_in[29];
    const float* out2_b   = (const float*)d_in[30];
    const float* W_c      = (const float*)d_in[31];
    const float* W_d      = (const float*)d_in[32];
    const float* W_xp     = (const float*)d_in[33];
    const float* b_p      = (const float*)d_in[34];
    float* out = (float*)d_out;

    // bump allocator in workspace
    char* base = (char*)d_ws;
    size_t off = 0;
    auto alloc = [&](size_t bytes) -> void* {
        off = (off + 255) & ~(size_t)255;
        void* p = base + off;
        off += bytes;
        return p;
    };
    auto pkElems = [](int Kp, int N) -> size_t {          // packed-B element count
        return (size_t)((N + 15) / 16) * (Kp / 32) * 512;
    };
    unsigned short* wfB  = (unsigned short*)alloc(pkElems(UNIP, G4) * 2);
    unsigned short* wzB  = (unsigned short*)alloc(pkElems(FENCP, G2) * 2);
    unsigned short* ufB  = (unsigned short*)alloc(pkElems(HP, G4) * 2);
    unsigned short* dwB  = (unsigned short*)alloc(pkElems(EP, G4) * 2);
    unsigned short* duB  = (unsigned short*)alloc(pkElems(HP, G4) * 2);
    unsigned short* wpT  = (unsigned short*)alloc(pkElems(HP, HH) * 2);
    unsigned short* wqT  = (unsigned short*)alloc(pkElems(HP, HH) * 2);
    unsigned short* wyT  = (unsigned short*)alloc(pkElems(HP, HH) * 2);
    unsigned short* o1T  = (unsigned short*)alloc(pkElems(HP, HH) * 2);
    unsigned short* wxT  = (unsigned short*)alloc(pkElems(FENCP, HH) * 2);
    unsigned short* wtT  = (unsigned short*)alloc(pkElems(H2P, HH) * 2);
    unsigned short* o2T  = (unsigned short*)alloc(pkElems(HP, VV) * 2);
    unsigned short* wordB= (unsigned short*)alloc((size_t)BB * SS * UNIP * 2);
    unsigned short* finB = (unsigned short*)alloc((size_t)BB * SS * FENCP * 2);
    unsigned short* decXB= (unsigned short*)alloc((size_t)BB * TT * EP * 2);
    float* XW    = (float*)alloc((size_t)BB * SS * G4 * 4);
    float* ZW    = (float*)alloc((size_t)BB * SS * G2 * 4);
    float* XdW   = (float*)alloc((size_t)BB * TT * G4 * 4);
    float* xwxp  = (float*)alloc((size_t)BB * TT * 4);
    float* en_out= (float*)alloc((size_t)BB * SS * HH * 4);
    unsigned short* enB = (unsigned short*)alloc((size_t)BB * SS * HP * 2);
    float* phi   = (float*)alloc((size_t)BB * SS * HH * 4);
    float* psi   = (float*)alloc((size_t)BB * SS * HH * 4);
    float* hbuf  = (float*)alloc((size_t)BB * HH * 4);
    float* cbuf  = (float*)alloc((size_t)BB * HH * 4);
    unsigned short* hB = (unsigned short*)alloc((size_t)BB * HP * 2);
    float* gbuf  = (float*)alloc((size_t)BB * G4 * 4);
    float* dq    = (float*)alloc((size_t)BB * HH * 4);
    float* dy    = (float*)alloc((size_t)BB * HH * 4);
    float* wat   = (float*)alloc((size_t)BB * SS * 4);
    float* ctx   = (float*)alloc((size_t)BB * HH * 4);
    unsigned short* cat = (unsigned short*)alloc((size_t)BB * H2P * 2);
    unsigned short* attB= (unsigned short*)alloc((size_t)BB * HP * 2);
    unsigned short* o1B = (unsigned short*)alloc((size_t)BB * HP * 2);
    float* fscratch = (float*)alloc((size_t)BB * HH * 4);
    float* logits   = (float*)alloc((size_t)BB * VV * 4);
    float* pg       = (float*)alloc((size_t)BB * 4);

    size_t used = (off + 255) & ~(size_t)255;
    if (used > ws_size) used = ws_size;   // defensive
    hipMemsetAsync(d_ws, 0, used, stream);  // zero scratch (guarantees pad columns = 0)

    // weights -> bf16, fragment-packed (pad folded in)
    auto pack = [&](const float* src, unsigned short* dst, int K, int N, int Kp, int tr) {
        const int KT = Kp / 32, NT = (N + 15) / 16;
        const int total = NT * KT * 512;
        pack_b<<<(total + 255) / 256, 256, 0, stream>>>(src, dst, K, N, KT, NT, tr);
    };
    pack(W_f,    wfB, UNI,  G4, UNIP,  0);
    pack(Wz,     wzB, FENC, G2, FENCP, 0);
    pack(U_f,    ufB, HH,   G4, HP,    0);
    pack(dec_W,  dwB, EE,   G4, EP,    0);
    pack(dec_U,  duB, HH,   G4, HP,    0);
    pack(Wp_w,   wpT, HH,   HH, HP,    1);
    pack(Wq_w,   wqT, HH,   HH, HP,    1);
    pack(Wy_w,   wyT, HH,   HH, HP,    1);
    pack(out1_w, o1T, HH,   HH, HP,    1);
    pack(Wx_w,   wxT, FENC, HH, FENCP, 1);
    pack(Wt_w,   wtT, 2*HH, HH, H2P,   1);
    pack(out2_w, o2T, HH,   VV, HP,    1);

    embed_concat<<<BB * SS, 256, 0, stream>>>(text, field, pos, rpos,
        enc_emb, field_emb, pos_emb, rpos_emb, wordB, finB);
    dec_embed<<<BB * TT, 256, 0, stream>>>(target, enc_emb, decXB);
    xwxp_kernel<<<(BB * TT * 32 + 255) / 256, 256, 0, stream>>>(target, enc_emb, W_xp, xwxp);

    auto gemm = [&](const unsigned short* A, int lda, const unsigned short* Bpk,
                    float* C, int ldc, const float* bias, const float* add, int ldadd,
                    unsigned short* Cb, int ldcb, int M, int N, int Kp, int act) {
        const int KT = Kp / 32, NT = (N + 15) / 16;
        dim3 grid((NT + 1) / 2, (M + 15) / 16);
        gemm_bf16<<<grid, 32, 0, stream>>>(A, lda, Bpk, C, ldc, bias,
                                           add, ldadd, Cb, ldcb, M, N, KT, NT, act);
    };

    // hoisted scan-invariant GEMMs (large, parallel)
    gemm(wordB, UNIP,  wfB, XW,  G4, nullptr, nullptr, 0, nullptr, 0, BB*SS, G4, UNIP, 0);
    gemm(finB,  FENCP, wzB, ZW,  G2, biasz,   nullptr, 0, nullptr, 0, BB*SS, G2, FENCP, 0);
    gemm(decXB, EP,    dwB, XdW, G4, nullptr, nullptr, 0, nullptr, 0, BB*TT, G4, EP, 0);

    const int gateBlocks = (BB * HH + 255) / 256;

    // ---------------- encoder scan ----------------
    for (int s = 0; s < SS; ++s) {
        gemm(hB, HP, ufB, gbuf, G4, bias_f, XW + (size_t)s * BB * G4, G4,
             nullptr, 0, BB, G4, HP, 0);
        enc_gate<<<gateBlocks, 256, 0, stream>>>(gbuf, ZW, hbuf, cbuf, hB, en_out, enB, s);
    }

    // attention keys (scan-invariant)
    gemm(enB,  HP,    wpT, phi, HH, Wp_b, nullptr, 0, nullptr, 0, BB*SS, HH, HP, 1);
    gemm(finB, FENCP, wxT, psi, HH, Wx_b, nullptr, 0, nullptr, 0, BB*SS, HH, FENCP, 1);

    // ---------------- decoder scan ----------------
    for (int t = 0; t < TT; ++t) {
        gemm(hB, HP, duB, gbuf, G4, dec_bias, XdW + (size_t)t * BB * G4, G4,
             nullptr, 0, BB, G4, HP, 0);
        dec_gate<<<gateBlocks, 256, 0, stream>>>(gbuf, hbuf, cbuf, hB);

        gemm(hB, HP, wqT, dq, HH, Wq_b, nullptr, 0, nullptr, 0, BB, HH, HP, 1);
        gemm(hB, HP, wyT, dy, HH, Wy_b, nullptr, 0, nullptr, 0, BB, HH, HP, 1);

        attn_kernel<<<BB, 128, 0, stream>>>(phi, psi, dq, dy, wat);
        ctx_kernel<<<gateBlocks, 256, 0, stream>>>(wat, en_out, hbuf, ctx, cat);

        gemm(cat,  H2P, wtT, fscratch, HH, Wt_b,   nullptr, 0, attB, HP, BB, HH, H2P, 1);
        gemm(attB, HP,  o1T, fscratch, HH, out1_b, nullptr, 0, o1B,  HP, BB, HH, HP, 0);
        gemm(o1B,  HP,  o2T, logits,   VV, out2_b, nullptr, 0, nullptr, 0, BB, VV, HP, 0);

        pgen_kernel<<<(BB * 32 + 255) / 256, 256, 0, stream>>>(ctx, hbuf, W_c, W_d, xwxp, b_p, pg, t);
        dist_kernel<<<BB, 256, 0, stream>>>(logits, pg, out, t);
        scatter_kernel<<<(BB * SS + 127) / 128, 128, 0, stream>>>(wat, pg, text, out, t);
    }
}
